// NeSVoR_36782099923508
// MI455X (gfx1250) — compile-verified, matrix operand order checked by reading, not run on hardware
//
#include <hip/hip_runtime.h>
#include <cstdint>
#include <cstddef>

// ---------------- problem constants ----------------
#define NPIX     131072
#define KSAMP    8
#define MSAMP    (NPIX * KSAMP)      // 1,048,576 sample points
#define NSLICES  128
#define TSIZE    524288u             // 2^19
#define TMASK    (TSIZE - 1u)
#define NLEV     16
#define HID      64
#define EMBED    16
#define P1       2654435761u
#define P2       805459861u

// output offsets (flat concat, all f32)
#define O_MEANI  0
#define O_TVAR   (NPIX)
#define O_VOL    (2 * NPIX)
#define O_X      (2 * NPIX + MSAMP)
#define O_LOGB   (2 * NPIX + MSAMP + 3 * MSAMP)

// floor(16 * 1.38^l)
__constant__ float RESF[NLEV] = {
    16.f, 22.f, 30.f, 42.f, 58.f, 80.f, 110.f, 152.f,
    210.f, 290.f, 400.f, 553.f, 763.f, 1053.f, 1453.f, 2005.f};

// ---------------- bf16 helpers ----------------
typedef __attribute__((ext_vector_type(16))) __bf16 v16bf;
typedef __attribute__((ext_vector_type(8)))  float  v8f;
typedef __attribute__((ext_vector_type(2)))  float  v2f;
typedef __attribute__((ext_vector_type(2)))  __bf16 v2bf;
typedef __attribute__((ext_vector_type(2)))  short  v2s;

union BFrag { v16bf v; unsigned short s[16]; uint4 q[2]; };

__device__ __forceinline__ float bf2f(unsigned short b) {
    union { unsigned u; float f; } v; v.u = ((unsigned)b) << 16;
    return v.f;
}
// packed f32x2 -> bf16x2 (low = a); vector convert -> single v_cvt_pk_bf16_f32
__device__ __forceinline__ unsigned cvt_pk_bf16(float a, float b) {
#if __has_builtin(__builtin_amdgcn_cvt_pk_bf16_f32)
    auto r = __builtin_amdgcn_cvt_pk_bf16_f32(a, b);
    return __builtin_bit_cast(unsigned, r);
#else
    v2f in = {a, b};
    v2bf r = __builtin_convertvector(in, v2bf);
    return __builtin_bit_cast(unsigned, r);
#endif
}
// packed ReLU on bf16x2 via integer ordering: one v_pk_max_i16
// (sign-magnitude: nonneg bf16 -> nonneg i16, negative bf16 -> negative i16)
__device__ __forceinline__ unsigned pk_relu_bf16x2(unsigned p) {
    v2s s = __builtin_bit_cast(v2s, p);
    v2s z = {0, 0};
    s = __builtin_elementwise_max(s, z);
    return __builtin_bit_cast(unsigned, s);
}

// ---------------- WMMA fragment load/store ----------------
// 16-bit A matrix 16x32 layout (ISA 7.12.2):
//   lanes 0-15 : row M=lane,    elems 0..7 -> K=0..7,  8..15 -> K=16..23
//   lanes 16-31: row M=lane-16, elems 0..7 -> K=8..15, 8..15 -> K=24..31
__device__ __forceinline__ v16bf load_A_lds(const unsigned short* rowbase, int ld, int kblock) {
    const int lane = threadIdx.x & 31;
    const int m    = lane & 15;
    const int hb   = lane >> 4;
    const unsigned short* p = rowbase + m * ld + kblock * 32 + hb * 8;
    BFrag u;
    u.q[0] = *(const uint4*)(p);
    u.q[1] = *(const uint4*)(p + 16);
    return u.v;
}

// B fragments pre-staged in LDS: frag f at wbase + f*512 ushorts, lane-contiguous 32B.
__device__ __forceinline__ v16bf load_B_lds(const unsigned short* wbase, int frag) {
    const int lane = threadIdx.x & 31;
    const unsigned short* p = wbase + frag * 512 + lane * 16;
    BFrag u;
    u.q[0] = *(const uint4*)(p);
    u.q[1] = *(const uint4*)(p + 8);
    return u.v;
}

// D layout: VGPR r, lane l -> (M = r + 8*(l>>4), N = l&15). Convert, packed ReLU, store.
template <bool RELU>
__device__ __forceinline__ void store_D(unsigned short* rowbase, int ld, int n0, v8f acc) {
    const int lane = threadIdx.x & 31;
    const int n    = n0 + (lane & 15);
    const int hb   = lane >> 4;
#pragma unroll
    for (int r = 0; r < 8; r += 2) {
        unsigned p = cvt_pk_bf16(acc[r], acc[r + 1]);
        if (RELU) p = pk_relu_bf16x2(p);
        rowbase[(r + 0 + 8 * hb) * ld + n] = (unsigned short)p;
        rowbase[(r + 1 + 8 * hb) * ld + n] = (unsigned short)(p >> 16);
    }
}

// Stage one layer's weights (Kin x Nout, row-major f32, row stride Wstride) into LDS
// B-fragment layout. Executed cooperatively by all 256 threads.
__device__ __forceinline__ void stage_frags(const float* __restrict__ W, int Wstride,
                                            int NK, int Kvalid, int NJ, int Nvalid,
                                            unsigned* __restrict__ dst /* u32 view */) {
    const int total = NK * NJ * 256;   // u32 words
    for (int i = threadIdx.x; i < total; i += 256) {
        const int f    = i >> 8;
        const int lane = (i >> 3) & 31;
        const int ep   = i & 7;
        const int j  = f / NK;
        const int kb = f - j * NK;
        const int n  = j * 16 + (lane & 15);
        const int hb = lane >> 4;
        const int e0 = ep * 2, e1 = e0 + 1;
        const int k0 = kb * 32 + (e0 & 7) + ((e0 & 8) << 1) + hb * 8;
        const int k1 = kb * 32 + (e1 & 7) + ((e1 & 8) << 1) + hb * 8;
        const float a = (k0 < Kvalid && n < Nvalid) ? W[k0 * Wstride + n] : 0.0f;
        const float b = (k1 < Kvalid && n < Nvalid) ? W[k1 * Wstride + n] : 0.0f;
        dst[f * 256 + lane * 8 + ep] = cvt_pk_bf16(a, b);
    }
}

// One layer tile: 16 samples x (NK*32) -> 16 x (NJ*16). Bias + optional ReLU.
// In-place safe (A fragments pre-loaded to VGPRs before any store).
template <int NK, int NJ, bool RELU>
__device__ __forceinline__ void mlp_layer(const unsigned short* inrow, int ldin,
                                          const unsigned short* wfrag,
                                          const float* bvec, int Nvalid,
                                          unsigned short* outrow, int ldout) {
    v16bf A[NK];
#pragma unroll
    for (int kb = 0; kb < NK; ++kb) A[kb] = load_A_lds(inrow, ldin, kb);
    const int lane = threadIdx.x & 31;
#pragma unroll
    for (int j = 0; j < NJ; ++j) {
        const int n = j * 16 + (lane & 15);
        const float bval = (n < Nvalid) ? bvec[n] : 0.0f;
        v8f acc;
#pragma unroll
        for (int r = 0; r < 8; ++r) acc[r] = bval;
#pragma unroll
        for (int kb = 0; kb < NK; ++kb) {
            v16bf B = load_B_lds(wfrag, j * NK + kb);
            acc = __builtin_amdgcn_wmma_f32_16x16x32_bf16(
                false, A[kb], false, B, (short)0, acc, false, false);
        }
        store_D<RELU>(outrow, ldout, j * 16, acc);
    }
}

// ---------------- kernel 0: per-slice softmax scaling + pose ----------------
__global__ void nesvor_prep(const float* __restrict__ logits,
                            const float* __restrict__ transforms,
                            float* __restrict__ ws) {
    __shared__ float red[NSLICES];
    const int s = threadIdx.x;
    float lg = logits[s];
    red[s] = lg; __syncthreads();
#pragma unroll
    for (int off = 64; off > 0; off >>= 1) {
        if (s < off) red[s] = fmaxf(red[s], red[s + off]);
        __syncthreads();
    }
    const float mx = red[0]; __syncthreads();
    float e = __expf(lg - mx);
    red[s] = e; __syncthreads();
#pragma unroll
    for (int off = 64; off > 0; off >>= 1) {
        if (s < off) red[s] += red[s + off];
        __syncthreads();
    }
    ws[NSLICES * 12 + s] = (float)NSLICES * e / red[0];

    // Rodrigues: R = cos(th) I + sin(th) K + (1-cos(th)) k k^T
    const float ax = transforms[s * 6 + 0];
    const float ay = transforms[s * 6 + 1];
    const float az = transforms[s * 6 + 2];
    float th = sqrtf(ax * ax + ay * ay + az * az);
    th = fmaxf(th, 1e-8f);
    const float inv = 1.0f / th;
    const float kx = ax * inv, ky = ay * inv, kz = az * inv;
    const float st = __sinf(th), ct = __cosf(th), oc = 1.0f - ct;
    float* R = ws + s * 12;
    R[0] = ct + oc * kx * kx;  R[1] = oc * kx * ky - st * kz;  R[2] = oc * kx * kz + st * ky;
    R[3] = oc * ky * kx + st * kz;  R[4] = ct + oc * ky * ky;  R[5] = oc * ky * kz - st * kx;
    R[6] = oc * kz * kx - st * ky;  R[7] = oc * kz * ky + st * kx;  R[8] = ct + oc * kz * kz;
    R[9]  = transforms[s * 6 + 3];
    R[10] = transforms[s * 6 + 4];
    R[11] = transforms[s * 6 + 5];
}

// ---------------- main fused kernel ----------------
// block = 256 threads (8 wave32) = 256 samples. Wave w owns LDS activation rows
// 32w..32w+31 (= its two 16-row WMMA tiles). Weights staged block-wide in LDS.
// Fragment map (1KB each): vol L1 0-3 | vol L2 4-11 | vol head 12-15 |
//                          bias L1 16-19 | bias L2 20-27 | var L1 28-31 | var L2 32-39
__global__ __launch_bounds__(256, 2) void nesvor_main(
    const float* __restrict__ pc, const float* __restrict__ sigma,
    const float* __restrict__ noise, const float* __restrict__ table,
    const float* __restrict__ volW1, const float* __restrict__ volb1,
    const float* __restrict__ volW2, const float* __restrict__ volb2,
    const float* __restrict__ volW3, const float* __restrict__ volb3,
    const float* __restrict__ biW1,  const float* __restrict__ bib1,
    const float* __restrict__ biW2,  const float* __restrict__ bib2,
    const float* __restrict__ biW3,  const float* __restrict__ bib3,
    const float* __restrict__ vaW1,  const float* __restrict__ vab1,
    const float* __restrict__ vaW2,  const float* __restrict__ vab2,
    const float* __restrict__ vaW3,  const float* __restrict__ vab3,
    const float* __restrict__ semb,  const float* __restrict__ lsv,
    const int*   __restrict__ sidx,  const float* __restrict__ wsRT,
    const float* __restrict__ wsC,   float* __restrict__ out) {

    extern __shared__ unsigned short smem[];
    unsigned short* wlds = smem;                    // 40 frags * 512 u16 = 40 KB
    unsigned short* enc  = smem + 40 * 512;         // 256 x 32 bf16 (16 KB)
    unsigned short* bufA = enc  + 256 * 32;         // 256 x 64 bf16 (32 KB)
    unsigned short* bufB = bufA + 256 * 64;         // 256 x 64 bf16 (32 KB)
    unsigned* wlds32 = (unsigned*)wlds;

    const int t   = threadIdx.x;
    const int gid = blockIdx.x * 256 + t;           // grid covers MSAMP exactly
    const int pix = gid >> 3;
    const int slc = sidx[pix];

    // -------- phase 0: stage GEMM weights into LDS fragments --------
    stage_frags(volW1, 64, 1, 32, 4, 64, wlds32 + 0  * 256);
    stage_frags(volW2, 64, 2, 64, 4, 64, wlds32 + 4  * 256);
    stage_frags(volW3, 17, 2, 64, 2, 17, wlds32 + 12 * 256);
    stage_frags(biW1,  64, 1, 24, 4, 64, wlds32 + 16 * 256);
    stage_frags(biW2,  64, 2, 64, 4, 64, wlds32 + 20 * 256);
    stage_frags(vaW1,  64, 1, 32, 4, 64, wlds32 + 28 * 256);
    stage_frags(vaW2,  64, 2, 64, 4, 64, wlds32 + 32 * 256);

    // -------- phase 1: transform + hash encode --------
    const float* RT = wsRT + slc * 12;
    float lc[3], x[3];
#pragma unroll
    for (int j = 0; j < 3; ++j)
        lc[j] = noise[(size_t)gid * 3 + j] * sigma[j] + pc[(size_t)pix * 3 + j];
#pragma unroll
    for (int i = 0; i < 3; ++i) {
        x[i] = RT[i * 3 + 0] * lc[0] + RT[i * 3 + 1] * lc[1] + RT[i * 3 + 2] * lc[2] + RT[9 + i];
        out[O_X + (size_t)gid * 3 + i] = x[i];
    }
    float xn[3];
#pragma unroll
    for (int i = 0; i < 3; ++i) xn[i] = fminf(fmaxf(x[i], 0.0f), 1.0f);

#pragma unroll 1
    for (int l = 0; l < NLEV; ++l) {
        // separable corner hash: {x, x+1}, {y*P1, y*P1+P1}, {z*P2, z*P2+P2}
        float w1[3]; unsigned c0[3];
#pragma unroll
        for (int d = 0; d < 3; ++d) {
            float xs = xn[d] * RESF[l];
            float fl = floorf(xs);
            w1[d] = xs - fl;
            c0[d] = (unsigned)(int)fl;
        }
        unsigned hx[2], hy[2], hz[2];
        hx[0] = c0[0];        hx[1] = hx[0] + 1u;
        hy[0] = c0[1] * P1;   hy[1] = hy[0] + P1;
        hz[0] = c0[2] * P2;   hz[1] = hz[0] + P2;
        const float wx0 = 1.f - w1[0], wy0 = 1.f - w1[1];
        float wz[2] = {1.f - w1[2], w1[2]};
        float pxy[4] = {wx0 * wy0, wx0 * w1[1], w1[0] * wy0, w1[0] * w1[1]};  // [ox*2+oy]
        float f0 = 0.f, f1 = 0.f;
#pragma unroll
        for (int c = 0; c < 8; ++c) {
            const int ox = (c >> 2) & 1, oy = (c >> 1) & 1, oz = c & 1;
            const unsigned h = (hx[ox] ^ hy[oy] ^ hz[oz]) & TMASK;
            const float2 f = *(const float2*)(table + ((size_t)l * TSIZE + h) * 2);
            const float wt = pxy[ox * 2 + oy] * wz[oz];
            f0 = fmaf(f.x, wt, f0);
            f1 = fmaf(f.y, wt, f1);
        }
        ((unsigned*)enc)[t * 16 + l] = cvt_pk_bf16(f0, f1);  // dims 2l, 2l+1
    }
    __syncthreads();

    // -------- phase 2a: volume MLP (WMMA): 32 -> 64 -> 64 -> 17(pad 32) --------
    const int wave = t >> 5;
#pragma unroll
    for (int ti = 0; ti < 2; ++ti) {
        const int rb = wave * 32 + ti * 16;
        mlp_layer<1, 4, true >(&enc[rb * 32],  32, wlds + 0  * 512, volb1, 64, &bufA[rb * 64], 64);
        mlp_layer<2, 4, true >(&bufA[rb * 64], 64, wlds + 4  * 512, volb2, 64, &bufB[rb * 64], 64);
        mlp_layer<2, 2, false>(&bufB[rb * 64], 64, wlds + 12 * 512, volb3, 17, &bufB[rb * 64], 64); // head, in-place
    }
    __syncthreads();

    // -------- phase 3a: read vol head, build bias/var inputs --------
    const float volume = bf2f(bufB[t * 64 + 0]);
    out[O_VOL + gid] = volume;
    // var input = [features(16) | emb(16)] -> bufA row; bias input = [low_enc(8) | emb(16) | 0(8)] -> enc row
#pragma unroll
    for (int e = 0; e < EMBED; ++e)
        bufA[t * 64 + e] = bufB[t * 64 + 1 + e];             // features, raw bf16 copy
#pragma unroll
    for (int e = 0; e < EMBED; e += 2) {
        const unsigned p = cvt_pk_bf16(semb[slc * EMBED + e], semb[slc * EMBED + e + 1]);
        ((unsigned*)bufA)[t * 32 + 8 + (e >> 1)] = p;        // emb -> bufA[16..31]
        ((unsigned*)enc)[t * 16 + 4 + (e >> 1)]  = p;        // emb -> enc[8..23]
    }
#pragma unroll
    for (int e = 0; e < 4; ++e) ((unsigned*)enc)[t * 16 + 12 + e] = 0;  // enc[24..31] = 0
    __syncthreads();

    // -------- phase 2b: bias MLP (K=24 zero-padded): 32 -> 64 -> 64 --------
#pragma unroll
    for (int ti = 0; ti < 2; ++ti) {
        const int rb = wave * 32 + ti * 16;
        mlp_layer<1, 4, true>(&enc[rb * 32],  32, wlds + 16 * 512, bib1, 64, &bufB[rb * 64], 64);
        mlp_layer<2, 4, true>(&bufB[rb * 64], 64, wlds + 20 * 512, bib2, 64, &bufB[rb * 64], 64); // in-place
    }
    __syncthreads();

    // -------- phase 3b: bias head (64 -> 1) --------
    float sb = bib3[0];
    for (int k = 0; k < HID; ++k) sb += bf2f(bufB[t * 64 + k]) * biW3[k];
    const float biasv = __expf(sb);
    out[O_LOGB + gid] = __logf(fmaxf(biasv, 1e-8f));
    __syncthreads();

    // -------- phase 2c: var MLP: 32 -> 64 -> 64 --------
#pragma unroll
    for (int ti = 0; ti < 2; ++ti) {
        const int rb = wave * 32 + ti * 16;
        mlp_layer<1, 4, true>(&bufA[rb * 64], 64, wlds + 28 * 512, vab1, 64, &bufB[rb * 64], 64);
        mlp_layer<2, 4, true>(&bufB[rb * 64], 64, wlds + 32 * 512, vab2, 64, &bufB[rb * 64], 64); // in-place
    }
    __syncthreads();

    // -------- phase 3c: var head + per-pixel reduction (K=8 lanes) --------
    float sv = vab3[0];
    for (int k = 0; k < HID; ++k) sv += bf2f(bufB[t * 64 + k]) * vaW3[k];
    const float varv = (sv > 20.0f) ? sv : log1pf(__expf(sv));

    float bv   = biasv * volume;
    float b2s2 = biasv * biasv * varv;
#pragma unroll
    for (int m = 1; m < 8; m <<= 1) {           // 8 consecutive lanes == one pixel
        bv   += __shfl_xor(bv,   m, 32);
        b2s2 += __shfl_xor(b2s2, m, 32);
    }
    if ((t & 7) == 0) {
        const float C = wsC[slc];
        out[O_MEANI + pix] = C * bv * 0.125f;
        out[O_TVAR  + pix] = C * C * b2s2 * 0.125f + __expf(lsv[slc]);
    }
}

// ---------------- host entry ----------------
extern "C" void kernel_launch(void* const* d_in, const int* in_sizes, int n_in,
                              void* d_out, int out_size, void* d_ws, size_t ws_size,
                              hipStream_t stream) {
    (void)in_sizes; (void)n_in; (void)out_size; (void)ws_size;
    const float* pc     = (const float*)d_in[0];
    const float* sigma  = (const float*)d_in[1];
    const float* noise  = (const float*)d_in[2];
    const float* table  = (const float*)d_in[3];
    const float* volW1  = (const float*)d_in[4];
    const float* volb1  = (const float*)d_in[5];
    const float* volW2  = (const float*)d_in[6];
    const float* volb2  = (const float*)d_in[7];
    const float* volW3  = (const float*)d_in[8];
    const float* volb3  = (const float*)d_in[9];
    const float* biW1   = (const float*)d_in[10];
    const float* bib1   = (const float*)d_in[11];
    const float* biW2   = (const float*)d_in[12];
    const float* bib2   = (const float*)d_in[13];
    const float* biW3   = (const float*)d_in[14];
    const float* bib3   = (const float*)d_in[15];
    const float* vaW1   = (const float*)d_in[16];
    const float* vab1   = (const float*)d_in[17];
    const float* vaW2   = (const float*)d_in[18];
    const float* vab2   = (const float*)d_in[19];
    const float* vaW3   = (const float*)d_in[20];
    const float* vab3   = (const float*)d_in[21];
    const float* semb   = (const float*)d_in[22];
    const float* logits = (const float*)d_in[23];
    const float* lsv    = (const float*)d_in[24];
    const float* trans  = (const float*)d_in[25];
    const int*   sidx   = (const int*)d_in[26];

    float* ws  = (float*)d_ws;                   // [128*12] R|t, then [128] C
    float* out = (float*)d_out;
    const float* wsRT = ws;
    const float* wsC  = ws + NSLICES * 12;

    nesvor_prep<<<1, NSLICES, 0, stream>>>(logits, trans, ws);

    const int smem_bytes = (40 * 512 + 256 * 32 + 2 * 256 * 64) * (int)sizeof(unsigned short); // 120 KB
    (void)hipFuncSetAttribute((const void*)nesvor_main,
                              hipFuncAttributeMaxDynamicSharedMemorySize, smem_bytes);
    nesvor_main<<<MSAMP / 256, 256, smem_bytes, stream>>>(
        pc, sigma, noise, table,
        volW1, volb1, volW2, volb2, volW3, volb3,
        biW1, bib1, biW2, bib2, biW3, bib3,
        vaW1, vab1, vaW2, vab2, vaW3, vab3,
        semb, lsv, sidx, wsRT, wsC, out);
}
